// Linear_73959336837249
// MI455X (gfx1250) — matrix operand order repaired and checked by reading, not compiled, not run on hardware
//
#include <hip/hip_runtime.h>
#include <cstdint>

// LoRA grouped linear for MI455X (gfx1250, wave32, WMMA).
// out = x @ W^T + scale_g * (x @ A_g^T) @ B_g^T   (per 4096-token segment g)
//
// Compute-bound at fp32 (137 GFLOP vs ~12us HBM floor).  Main GEMM uses the
// bf16x3 split scheme on V_WMMA_F32_16X16X32_BF16 (8x FLOPs/instruction vs the
// f32 WMMA, 3x the work => ~2.7x net, ~16 mantissa bits + fp32 accumulate),
// with register double-buffering of the K-tiles.  Pure-copy stages (kernel 1
// tiles, rank-16 tiles) use GLOBAL_LOAD_ASYNC_TO_LDS_B128 (ASYNCcnt).
// The rank-16 LoRA tail and `low` kernel stay on exact-fp32 WMMA 16x16x4.

typedef __attribute__((ext_vector_type(2)))  float  v2f;
typedef __attribute__((ext_vector_type(8)))  float  v8f;
typedef __attribute__((ext_vector_type(4)))  int    v4i;
typedef __bf16 bf16;
typedef __attribute__((ext_vector_type(4)))  __bf16 v4bf;
typedef __attribute__((ext_vector_type(8)))  __bf16 v8bf;
typedef __attribute__((ext_vector_type(16))) __bf16 v16bf;

typedef __attribute__((address_space(1))) v4i gv4i;   // global int4
typedef __attribute__((address_space(3))) v4i lv4i;   // LDS int4

#ifndef __has_builtin
#define __has_builtin(x) 0
#endif

#define TOKENS 16384
#define D_IN   2048
#define D_OUT  2048
#define RANK   16
#define SEG_BLOCKS 32         // 4096 tokens per adapter / BM

#define BM 128
#define BN 128
#define BK 32                 // K per LDS stage == K per bf16 WMMA
#define SA 40                 // bf16 LDS row stride: 80B rows (16B aligned), conflict-free frags
#define FK 32                 // fp32 K-tile for the low kernel
#define XST (FK + 4)          // 36-float stride (low kernel fp32 tiles)
#define RST (RANK + 4)        // 20-float stride for fp32 rank tiles

// ---------------- async global->LDS (16B per lane, ASYNCcnt) ----------------
__device__ __forceinline__ void async_g2l_b128(void* lds_dst, const void* gsrc) {
#if __has_builtin(__builtin_amdgcn_global_load_async_to_lds_b128)
    __builtin_amdgcn_global_load_async_to_lds_b128((gv4i*)gsrc, (lv4i*)lds_dst, 0, 0);
#else
    asm volatile("global_load_async_to_lds_b128 %0, %1, off"
                 :: "v"((uint32_t)(uintptr_t)(lv4i*)lds_dst),
                    "v"(gsrc)
                 : "memory");
#endif
}

__device__ __forceinline__ void wait_async0() {
#if __has_builtin(__builtin_amdgcn_s_wait_asynccnt)
    __builtin_amdgcn_s_wait_asynccnt(0);
#else
    asm volatile("s_wait_asynccnt 0x0" ::: "memory");
#endif
}

// ---------------- fp32 WMMA (16x16x4) helpers: exact path ----------------
__device__ __forceinline__ v2f frag2(const float* t, int rowBase, int stride, int k0) {
    const int lane = threadIdx.x & 31;
    const int r = rowBase + (lane & 15);
    const int k = k0 + ((lane >> 4) << 1);
    return *(const v2f*)(t + r * stride + k);
}

__device__ __forceinline__ v8f wmma4(v2f a, v2f b, v8f c) {
    return __builtin_amdgcn_wmma_f32_16x16x4_f32(false, a, false, b, (short)0, c, false, false);
}

// ---------------- bf16 WMMA (16x16x32) helpers: fast path ----------------
__device__ __forceinline__ v8f wmma32(v16bf a, v16bf b, v8f c) {
    return __builtin_amdgcn_wmma_f32_16x16x32_bf16(false, a, false, b, (short)0, c, false, false);
}

// A-matrix 16x32 bf16 fragment (ISA 7.12.2): lane m = lane&15,
// kb = 8*(lane>=16); K-chunks [kb, kb+8) and [kb+16, kb+24).
__device__ __forceinline__ v16bf fragA(const bf16* tile, int rowBase) {
    const int lane = threadIdx.x & 31;
    const bf16* p = tile + (rowBase + (lane & 15)) * SA + ((lane >> 4) << 3);
    v8bf c0 = *(const v8bf*)(p);
    v8bf c1 = *(const v8bf*)(p + 16);
    return __builtin_shufflevector(c0, c1, 0,1,2,3,4,5,6,7,8,9,10,11,12,13,14,15);
}

// B-matrix 32x16 bf16 fragment: lane n = lane&15, kb = 8*(lane>=16);
// K-chunks [kb, kb+8) and [kb+16, kb+24) of row n of the (K-contiguous) W tile.
__device__ __forceinline__ v16bf fragB(const bf16* tile, int rowBase) {
    const int lane = threadIdx.x & 31;
    const bf16* p = tile + (rowBase + (lane & 15)) * SA + ((lane >> 4) << 3);
    v8bf c0 = *(const v8bf*)(p);
    v8bf c1 = *(const v8bf*)(p + 16);
    return __builtin_shufflevector(c0, c1, 0,1,2,3,4,5,6,7,8,9,10,11,12,13,14,15);
}

// Split one float4 into hi/lo bf16 quads and store to the two LDS tiles.
__device__ __forceinline__ void split_store(float4 v, bf16* hi, bf16* lo) {
    float f0 = v.x, f1 = v.y, f2 = v.z, f3 = v.w;
    v4bf h, l;
    bf16 b;
    b = (bf16)f0; h[0] = b; l[0] = (bf16)(f0 - (float)b);
    b = (bf16)f1; h[1] = b; l[1] = (bf16)(f1 - (float)b);
    b = (bf16)f2; h[2] = b; l[2] = (bf16)(f2 - (float)b);
    b = (bf16)f3; h[3] = b; l[3] = (bf16)(f3 - (float)b);
    *(v4bf*)hi = h;
    *(v4bf*)lo = l;
}

// ---------------------------------------------------------------------------
// Kernel 1: low[t, r] = scalings[g] * sum_k x[t,k] * lora_a[g][r][k]
// Exact fp32 WMMA; memory-bound; tiles staged with async global->LDS.
// ---------------------------------------------------------------------------
__global__ __launch_bounds__(256)
void lora_low_kernel(const float* __restrict__ x,
                     const float* __restrict__ lora_a,
                     const float* __restrict__ scalings,
                     float* __restrict__ low)
{
    __shared__ float Xs[BM * XST];
    __shared__ float As[RANK * XST];

    const int tid  = threadIdx.x;
    const int lane = tid & 31;
    const int wave = tid >> 5;
    const int m0   = blockIdx.x * BM;
    const int g    = blockIdx.x / SEG_BLOCKS;

    v8f acc = (v8f){0.f, 0.f, 0.f, 0.f, 0.f, 0.f, 0.f, 0.f};

    const int r = tid >> 3;   // 0..31
    const int c = tid & 7;    // float4 column
    for (int kt = 0; kt < D_IN; kt += FK) {
        __syncthreads();
        #pragma unroll
        for (int p = 0; p < 4; ++p) {
            const int row = r + p * 32;
            async_g2l_b128(Xs + row * XST + c * 4,
                           x + (size_t)(m0 + row) * D_IN + kt + c * 4);
        }
        if (tid < RANK * 8) {
            const int ar = tid >> 3;
            async_g2l_b128(As + ar * XST + c * 4,
                           lora_a + ((size_t)g * RANK + ar) * D_IN + kt + c * 4);
        }
        wait_async0();
        __syncthreads();
        #pragma unroll
        for (int ks = 0; ks < FK; ks += 4) {
            v2f a = frag2(Xs, wave * 16, XST, ks);
            v2f b = frag2(As, 0, XST, ks);
            acc = wmma4(a, b, acc);
        }
    }

    const float s  = scalings[g];
    const int mrow = m0 + wave * 16 + ((lane >> 4) << 3);
    const int col  = lane & 15;
    #pragma unroll
    for (int e = 0; e < 8; ++e)
        low[(size_t)(mrow + e) * RANK + col] = acc[e] * s;
}

// ---------------------------------------------------------------------------
// Kernel 2: out = x @ W^T (bf16x3 on v_wmma_f32_16x16x32_bf16, register
// double-buffered K-tiles) + rank-16 tail: low @ B_g^T (exact fp32 WMMA,
// async-staged before the loop, waited after it).
// 128x128 block tile, 8 waves as 4(m) x 2(n); each wave owns 32x64.
// ---------------------------------------------------------------------------
__global__ __launch_bounds__(256)
void lora_gemm_kernel(const float* __restrict__ x,
                      const float* __restrict__ W,
                      const float* __restrict__ lora_b,
                      const float* __restrict__ low,
                      float* __restrict__ out)
{
    __shared__ bf16  Xh[BM * SA];
    __shared__ bf16  Xl[BM * SA];
    __shared__ bf16  Wh[BN * SA];
    __shared__ bf16  Wl[BN * SA];
    __shared__ float Ls[BM * RST];   // low tile  [128][16] fp32
    __shared__ float Bs[BN * RST];   // B_g tile  [128][16] fp32

    const int tid  = threadIdx.x;
    const int lane = tid & 31;
    const int wave = tid >> 5;
    const int wm   = wave >> 1;      // 0..3 -> 32-row slab
    const int wn   = wave & 1;       // 0..1 -> 64-col slab
    const int m0   = blockIdx.x * BM;
    const int n0   = blockIdx.y * BN;
    const int g    = blockIdx.x / SEG_BLOCKS;

    // Kick off async staging of the rank-16 fp32 operands; they are only
    // consumed after the main K loop, so the copy overlaps the entire GEMM.
    {
        const int rr = tid >> 2;     // 0..63
        const int cc = tid & 3;      // float4 column of 16 floats
        #pragma unroll
        for (int p = 0; p < 2; ++p) {
            const int row = rr + p * 64;
            async_g2l_b128(Ls + row * RST + cc * 4,
                           low + (size_t)(m0 + row) * RANK + cc * 4);
            async_g2l_b128(Bs + row * RST + cc * 4,
                           lora_b + ((size_t)g * D_OUT + n0 + row) * RANK + cc * 4);
        }
    }

    v8f acc[2][4];
    #pragma unroll
    for (int i = 0; i < 2; ++i)
        #pragma unroll
        for (int j = 0; j < 4; ++j)
            acc[i][j] = (v8f){0.f, 0.f, 0.f, 0.f, 0.f, 0.f, 0.f, 0.f};

    const int r = tid >> 3;   // 0..31
    const int c = tid & 7;    // float4 column within BK

    // Prologue: fetch K-tile 0 into registers.
    float4 vx[4], vw[4];
    #pragma unroll
    for (int p = 0; p < 4; ++p) {
        const int row = r + p * 32;
        vx[p] = *(const float4*)(x + (size_t)(m0 + row) * D_IN + c * 4);
        vw[p] = *(const float4*)(W + (size_t)(n0 + row) * D_IN + c * 4);
    }

    for (int kt = 0; kt < D_IN; kt += BK) {
        __syncthreads();              // previous compute done -> LDS reusable
        #pragma unroll
        for (int p = 0; p < 4; ++p) {
            const int row = r + p * 32;
            split_store(vx[p], Xh + row * SA + c * 4, Xl + row * SA + c * 4);
            split_store(vw[p], Wh + row * SA + c * 4, Wl + row * SA + c * 4);
        }
        __syncthreads();

        // Prefetch the next K-tile; overlaps the WMMA block below.
        if (kt + BK < D_IN) {
            #pragma unroll
            for (int p = 0; p < 4; ++p) {
                const int row = r + p * 32;
                vx[p] = *(const float4*)(x + (size_t)(m0 + row) * D_IN + kt + BK + c * 4);
                vw[p] = *(const float4*)(W + (size_t)(n0 + row) * D_IN + kt + BK + c * 4);
            }
        }

        v16bf ah[2], al[2], bh[4], bl[4];
        #pragma unroll
        for (int i = 0; i < 2; ++i) {
            ah[i] = fragA(Xh, wm * 32 + i * 16);
            al[i] = fragA(Xl, wm * 32 + i * 16);
        }
        #pragma unroll
        for (int j = 0; j < 4; ++j) {
            bh[j] = fragB(Wh, wn * 64 + j * 16);
            bl[j] = fragB(Wl, wn * 64 + j * 16);
        }
        // Three sweeps; consecutive WMMAs never chain on the same accumulator.
        #pragma unroll
        for (int i = 0; i < 2; ++i)
            #pragma unroll
            for (int j = 0; j < 4; ++j)
                acc[i][j] = wmma32(ah[i], bh[j], acc[i][j]);
        #pragma unroll
        for (int i = 0; i < 2; ++i)
            #pragma unroll
            for (int j = 0; j < 4; ++j)
                acc[i][j] = wmma32(ah[i], bl[j], acc[i][j]);
        #pragma unroll
        for (int i = 0; i < 2; ++i)
            #pragma unroll
            for (int j = 0; j < 4; ++j)
                acc[i][j] = wmma32(al[i], bh[j], acc[i][j]);
    }

    // Rank-16 LoRA K-tail: exact fp32 WMMA from the async-staged rank tiles.
    wait_async0();
    __syncthreads();
    #pragma unroll
    for (int ks = 0; ks < RANK; ks += 4) {
        v2f af[2], bf[4];
        #pragma unroll
        for (int i = 0; i < 2; ++i) af[i] = frag2(Ls, wm * 32 + i * 16, RST, ks);
        #pragma unroll
        for (int j = 0; j < 4; ++j) bf[j] = frag2(Bs, wn * 64 + j * 16, RST, ks);
        #pragma unroll
        for (int i = 0; i < 2; ++i)
            #pragma unroll
            for (int j = 0; j < 4; ++j)
                acc[i][j] = wmma4(af[i], bf[j], acc[i][j]);
    }

    // Epilogue: C/D layout -> lane (n = lane&15, m = vgpr + 8*(lane>=16)).
    const int lm = (lane >> 4) << 3;
    const int ln = lane & 15;
    #pragma unroll
    for (int i = 0; i < 2; ++i) {
        const int mb = m0 + wm * 32 + i * 16 + lm;
        #pragma unroll
        for (int j = 0; j < 4; ++j) {
            const int nn = n0 + wn * 64 + j * 16 + ln;
            #pragma unroll
            for (int e = 0; e < 8; ++e)
                out[(size_t)(mb + e) * D_OUT + nn] = acc[i][j][e];
        }
    }
}

extern "C" void kernel_launch(void* const* d_in, const int* in_sizes, int n_in,
                              void* d_out, int out_size, void* d_ws, size_t ws_size,
                              hipStream_t stream) {
    (void)in_sizes; (void)n_in; (void)out_size; (void)ws_size;
    const float* x        = (const float*)d_in[0];   // [16384, 2048]
    const float* W        = (const float*)d_in[1];   // [2048, 2048]
    const float* lora_a   = (const float*)d_in[2];   // [4, 16, 2048]
    const float* lora_b   = (const float*)d_in[3];   // [4, 2048, 16]
    const float* scalings = (const float*)d_in[4];   // [4]
    float* out = (float*)d_out;                      // [16384, 2048]
    float* low = (float*)d_ws;                       // [16384, 16] scratch (1 MB)

    lora_low_kernel<<<TOKENS / BM, 256, 0, stream>>>(x, lora_a, scalings, low);

    dim3 grid(TOKENS / BM, D_OUT / BN);              // (128, 16)
    lora_gemm_kernel<<<grid, 256, 0, stream>>>(x, W, lora_b, low, out);
}
